// MDTA_2087354106138
// MI455X (gfx1250) — matrix operand order repaired and checked by reading, not compiled, not run on hardware
//
#include <hip/hip_runtime.h>
#include <hip/hip_bf16.h>
#include <math.h>

#define B_   2
#define C_   192
#define H_   256
#define W_   256
#define NH_  4
#define HD_  48          // C_/NH_
#define HW_  (H_*W_)     // 65536
#define C3_  (3*C_)      // 576
#define KSPLIT_ 128      // split-K chunks for q@k^T

#define GN_   256        // GEMM block N tile (4 waves x 64 cols)
#define GK_   32         // GEMM K step (one bf16 WMMA depth)
#define BSTR_ 40         // LDS row stride in bf16 elems (80 B: 16B-aligned, conflict-free)

typedef __attribute__((ext_vector_type(16))) __bf16 v16bf;
typedef __attribute__((ext_vector_type(8)))  float  v8f;
typedef __attribute__((ext_vector_type(2)))  float  v2f;

// ---------------------------------------------------------------------------
// Stage one 32K x 256N f32 tile of X into LDS, transposed to [n][k] bf16
// (padded stride). 8192 floats; each of 128 threads moves 16 float4's.
// ---------------------------------------------------------------------------
__device__ __forceinline__ void stage_tile(const float* __restrict__ Xb,
                                           int kb, int bn, int Ns, int K,
                                           int t, __bf16* __restrict__ dst)
{
#pragma unroll
    for (int i = 0; i < 16; ++i) {
        const int flat = (i * 128 + t) * 4;     // 0..8188
        const int kk   = flat >> 8;             // 0..31
        const int nn   = flat & 255;
        const float* src = Xb + (size_t)(kb + kk) * Ns + bn + nn;
        if (i == 0 && kb + GK_ < K)
            __builtin_prefetch(src + (size_t)GK_ * Ns, 0, 1);  // global_prefetch_b8
        const float4 v = *(const float4*)src;
        dst[(nn + 0) * BSTR_ + kk] = (__bf16)v.x;
        dst[(nn + 1) * BSTR_ + kk] = (__bf16)v.y;
        dst[(nn + 2) * BSTR_ + kk] = (__bf16)v.z;
        dst[(nn + 3) * BSTR_ + kk] = (__bf16)v.w;
    }
}

// ---------------------------------------------------------------------------
// Pointwise-conv GEMM: Y[b][m][n] = sum_k Wm[m][k] * X[b][k][n]
// Block = 128 threads (4 waves). Block tile = 16(M) x 256(N); wave w owns
// 64 columns (4 accumulators). Double-buffered LDS staging overlaps the
// HBM stream with WMMA issue; all 4 B fragments are loaded before the 4
// back-to-back WMMAs so a single dscnt wait covers them.
// M%16==0, K%32==0, Ns%256==0.
// ---------------------------------------------------------------------------
__global__ void __launch_bounds__(128)
gemm_bf16_wmma(const float* __restrict__ Wm,  // [M,K] row-major
               const float* __restrict__ X,   // [B][K,Ns]
               float* __restrict__ Y,         // [B][M,Ns]
               int M, int K, int Ns)
{
    const int t    = threadIdx.x;
    const int lane = t & 31;
    const int wv   = t >> 5;           // wave id 0..3
    const int h    = lane >> 4;        // half
    const int l    = lane & 15;
    const int bn   = blockIdx.x * GN_;
    const int tm   = blockIdx.y * 16;
    const int b    = blockIdx.z;

    const float* Xb = X + (size_t)b * K * Ns;
    float*       Yb = Y + (size_t)b * M * Ns;

    __shared__ __bf16 Bs[2][GN_ * BSTR_];   // double-buffered [n][k] bf16 tiles

    v8f acc[4] = {};
    const float* arow = Wm + (size_t)(tm + l) * K;

    stage_tile(Xb, 0, bn, Ns, K, t, Bs[0]);
    __syncthreads();

    for (int kb = 0; kb < K; kb += GK_) {
        const int cur = (kb / GK_) & 1;

        // ---- stage next K-slab into the other buffer (overlaps with WMMA)
        if (kb + GK_ < K)
            stage_tile(Xb, kb + GK_, bn, Ns, K, t, Bs[cur ^ 1]);

        // ---- A fragment straight from global (rows contiguous, L2-hot)
        v16bf afrag;
        const int ka0 = kb + h * 16;       // elements 0..7  -> K = ka0 + e
        const int ka1 = kb + h * 16 + 8;   // elements 8..15 -> K = ka1 + (e-8)
#pragma unroll
        for (int e = 0; e < 8; ++e) {
            afrag[e]     = (__bf16)arow[ka0 + e];
            afrag[e + 8] = (__bf16)arow[ka1 + e];
        }

        // ---- load all 4 B fragments (8x ds_load_b128), then 4 WMMAs
        v16bf bfr[4];
#pragma unroll
        for (int j = 0; j < 4; ++j) {
            const int cn = wv * 64 + j * 16 + l;               // column in block tile
            const __bf16* bp = &Bs[cur][cn * BSTR_ + h * 16];  // K = h*16 + e
#pragma unroll
            for (int e = 0; e < 16; ++e) bfr[j][e] = bp[e];
        }
#pragma unroll
        for (int j = 0; j < 4; ++j)
            acc[j] = __builtin_amdgcn_wmma_f32_16x16x32_bf16(
                         false, afrag, false, bfr[j], (short)0, acc[j], false, false);

        __syncthreads();   // covers read-before-overwrite and write-before-read
    }

#pragma unroll
    for (int j = 0; j < 4; ++j) {
#pragma unroll
        for (int r = 0; r < 8; ++r) {
            const int m = tm + r + 8 * h;                 // C/D layout
            const int n = bn + wv * 64 + j * 16 + l;
            Yb[(size_t)m * Ns + n] = acc[j][r];
        }
    }
}

// ---------------------------------------------------------------------------
// Depthwise 3x3, groups = 3C, SAME padding.
// ---------------------------------------------------------------------------
__global__ void dwconv3x3(const float* __restrict__ in,
                          const float* __restrict__ w,
                          float* __restrict__ out)
{
    const size_t idx = (size_t)blockIdx.x * blockDim.x + threadIdx.x;
    const int x  = (int)(idx % W_);
    const int y  = (int)((idx / W_) % H_);
    const int ch = (int)((idx / HW_) % C3_);

    const float* wp = w + (size_t)ch * 9;
    const float* ip = in + (idx - (size_t)x - (size_t)y * W_);  // plane base

    float s = 0.f;
#pragma unroll
    for (int dy = -1; dy <= 1; ++dy) {
        const int yy = y + dy;
        if (yy < 0 || yy >= H_) continue;
#pragma unroll
        for (int dx = -1; dx <= 1; ++dx) {
            const int xx = x + dx;
            if (xx < 0 || xx >= W_) continue;
            s += wp[(dy + 1) * 3 + (dx + 1)] * ip[(size_t)yy * W_ + xx];
        }
    }
    out[idx] = s;
}

// ---------------------------------------------------------------------------
// Per-row inverse L2 norm over the spatial axis for q (ch 0..C-1) and
// k (ch C..2C-1) channels of the depthwise output.
// ---------------------------------------------------------------------------
__global__ void rownorm(const float* __restrict__ qkv, float* __restrict__ invn)
{
    const int ch = blockIdx.x;          // 0 .. 2C-1
    const int b  = blockIdx.y;
    const float* row = qkv + ((size_t)b * C3_ + ch) * HW_;

    float s = 0.f;
    for (int i = threadIdx.x; i < HW_; i += blockDim.x) {
        const float v = row[i];
        s += v * v;
    }
    __shared__ float red[256];
    red[threadIdx.x] = s;
    __syncthreads();
    for (int off = 128; off > 0; off >>= 1) {
        if ((int)threadIdx.x < off) red[threadIdx.x] += red[threadIdx.x + off];
        __syncthreads();
    }
    if (threadIdx.x == 0) {
        const float n = sqrtf(red[0]);
        invn[(size_t)b * (2 * C_) + ch] = 1.0f / fmaxf(n, 1e-12f);
    }
}

__global__ void zero_f32(float* __restrict__ p, int n)
{
    const int i = blockIdx.x * blockDim.x + threadIdx.x;
    if (i < n) p[i] = 0.f;
}

// ---------------------------------------------------------------------------
// attn_raw[bh][c][d] += sum_{n in K-chunk} q[c][n] * k[d][n]
// One wave computes the FULL 48x48 head block for one K-chunk: 3x3 tiles of
// v_wmma_f32_16x16x4_f32 (9 WMMAs per 6 float2 loads), atomically reduced.
// ---------------------------------------------------------------------------
__global__ void __launch_bounds__(32)
attn_qk_wmma(const float* __restrict__ qkv, float* __restrict__ attn_raw)
{
    const int lane = threadIdx.x & 31;
    const int h    = lane >> 4;
    const int l    = lane & 15;
    const int bh   = blockIdx.y;
    const int b    = bh / NH_;
    const int hd   = bh % NH_;
    const int KCH  = HW_ / KSPLIT_;
    const int k0   = blockIdx.x * KCH;

    const float* qbase = qkv + ((size_t)b * C3_ + (size_t)hd * HD_) * HW_;
    const float* kbase = qkv + ((size_t)b * C3_ + C_ + (size_t)hd * HD_) * HW_;

    const float* qr[3];
    const float* kr[3];
#pragma unroll
    for (int i = 0; i < 3; ++i) {
        qr[i] = qbase + (size_t)(i * 16 + l) * HW_;   // f32 A 16x4: row tm+l
        kr[i] = kbase + (size_t)(i * 16 + l) * HW_;   // f32 B 4x16: col tn+l
    }

    v8f acc[3][3] = {};
    for (int kk = k0; kk < k0 + KCH; kk += 4) {
        const int off = kk + 2 * h;                   // lane half holds K={2h,2h+1}
        v2f a[3], bb[3];
#pragma unroll
        for (int i = 0; i < 3; ++i) {
            a[i].x  = qr[i][off];  a[i].y  = qr[i][off + 1];
            bb[i].x = kr[i][off];  bb[i].y = kr[i][off + 1];
        }
#pragma unroll
        for (int i = 0; i < 3; ++i)
#pragma unroll
            for (int j = 0; j < 3; ++j)
                acc[i][j] = __builtin_amdgcn_wmma_f32_16x16x4_f32(
                                false, a[i], false, bb[j], (short)0, acc[i][j],
                                false, false);
    }

#pragma unroll
    for (int i = 0; i < 3; ++i)
#pragma unroll
        for (int j = 0; j < 3; ++j)
#pragma unroll
            for (int r = 0; r < 8; ++r) {
                const int c = i * 16 + r + 8 * h;
                const int d = j * 16 + l;
                atomicAdd(&attn_raw[(size_t)bh * HD_ * HD_ + (size_t)c * HD_ + d],
                          acc[i][j][r]);
            }
}

// ---------------------------------------------------------------------------
// Apply q/k inverse norms + temperature, then softmax over d (48 wide).
// ---------------------------------------------------------------------------
__global__ void attn_softmax(const float* __restrict__ raw,
                             const float* __restrict__ invn,
                             const float* __restrict__ temp,
                             float* __restrict__ sm)
{
    const int c  = blockIdx.x;
    const int bh = blockIdx.y;
    const int b  = bh / NH_;
    const int hh = bh % NH_;
    const int t  = threadIdx.x;

    __shared__ float sv[HD_];
    __shared__ float ev[HD_];

    const float iq = invn[(size_t)b * 2 * C_ + hh * HD_ + c];
    const float ts = temp[hh];

    float s = 0.f;
    if (t < HD_) {
        const float ik = invn[(size_t)b * 2 * C_ + C_ + hh * HD_ + t];
        s = raw[(size_t)bh * HD_ * HD_ + (size_t)c * HD_ + t] * iq * ik * ts;
        sv[t] = s;
    }
    __syncthreads();
    float mx = -3.4e38f;
    for (int i = 0; i < HD_; ++i) mx = fmaxf(mx, sv[i]);
    float e = 0.f;
    if (t < HD_) { e = expf(s - mx); ev[t] = e; }
    __syncthreads();
    float sum = 0.f;
    for (int i = 0; i < HD_; ++i) sum += ev[i];
    if (t < HD_) sm[(size_t)bh * HD_ * HD_ + (size_t)c * HD_ + t] = e / sum;
}

// ---------------------------------------------------------------------------
// out[b][h*48+c][n] = sum_d attn[bh][c][d] * v[b][2C + h*48 + d][n]
// ---------------------------------------------------------------------------
__global__ void attn_v(const float* __restrict__ sm,
                       const float* __restrict__ qkv,
                       float* __restrict__ outbuf)
{
    const int t  = threadIdx.x;
    const int n  = blockIdx.x * 256 + t;
    const int bh = blockIdx.y;
    const int b  = bh / NH_;
    const int hh = bh % NH_;

    __shared__ float A[HD_ * HD_];
    for (int i = t; i < HD_ * HD_; i += 256)
        A[i] = sm[(size_t)bh * HD_ * HD_ + i];
    __syncthreads();

    const float* vbase = qkv + ((size_t)b * C3_ + 2 * C_ + (size_t)hh * HD_) * HW_ + n;

    float acc[HD_];
#pragma unroll
    for (int c = 0; c < HD_; ++c) acc[c] = 0.f;

    for (int d = 0; d < HD_; ++d) {
        const float vv = vbase[(size_t)d * HW_];
#pragma unroll
        for (int c = 0; c < HD_; ++c) acc[c] += A[c * HD_ + d] * vv;
    }

    float* ob = outbuf + ((size_t)b * C_ + (size_t)hh * HD_) * HW_ + n;
#pragma unroll
    for (int c = 0; c < HD_; ++c) ob[(size_t)c * HW_] = acc[c];
}

// ---------------------------------------------------------------------------
extern "C" void kernel_launch(void* const* d_in, const int* in_sizes, int n_in,
                              void* d_out, int out_size, void* d_ws, size_t ws_size,
                              hipStream_t stream)
{
    (void)in_sizes; (void)n_in; (void)out_size; (void)ws_size;

    const float* x      = (const float*)d_in[0];   // [B,C,H,W]
    const float* qkv_w  = (const float*)d_in[1];   // [3C,C]
    const float* dw_w   = (const float*)d_in[2];   // [3C,1,3,3]
    const float* proj_w = (const float*)d_in[3];   // [C,C]
    const float* temp   = (const float*)d_in[4];   // [NH]
    float* out = (float*)d_out;                    // [B,C,H,W]

    float* ws = (float*)d_ws;
    const size_t qkv_elems = (size_t)B_ * C3_ * HW_;      // 75,497,472 floats
    float* qkv_raw  = ws;                                  // [B,3C,HW]; reused for attn output [B,C,HW]
    float* qkv_dw   = ws + qkv_elems;                      // [B,3C,HW]
    float* attn_raw = ws + 2 * qkv_elems;                  // [B*NH,48,48]
    float* attn_sm  = attn_raw + (size_t)B_ * NH_ * HD_ * HD_;
    float* invn     = attn_sm  + (size_t)B_ * NH_ * HD_ * HD_;   // [B, 2C]

    // 1) qkv = W_qkv @ x           (WMMA bf16, f32 accum, double-buffered LDS)
    gemm_bf16_wmma<<<dim3(HW_ / GN_, C3_ / 16, B_), 128, 0, stream>>>(
        qkv_w, x, qkv_raw, C3_, C_, HW_);

    // 2) depthwise 3x3, groups = 3C
    dwconv3x3<<<(int)(((size_t)B_ * C3_ * HW_) / 256), 256, 0, stream>>>(
        qkv_raw, dw_w, qkv_dw);

    // 3) inverse L2 norms of q/k rows
    rownorm<<<dim3(2 * C_, B_), 256, 0, stream>>>(qkv_dw, invn);

    // 4) attn_raw = q @ k^T        (WMMA f32 16x16x4, split-K + atomics)
    zero_f32<<<(B_ * NH_ * HD_ * HD_ + 255) / 256, 256, 0, stream>>>(
        attn_raw, B_ * NH_ * HD_ * HD_);
    attn_qk_wmma<<<dim3(KSPLIT_, B_ * NH_), 32, 0, stream>>>(qkv_dw, attn_raw);

    // 5) scale + softmax
    attn_softmax<<<dim3(HD_, B_ * NH_), 64, 0, stream>>>(attn_raw, invn, temp, attn_sm);

    // 6) out = attn @ v   (reuse qkv_raw region as [B,C,HW] buffer)
    attn_v<<<dim3(HW_ / 256, B_ * NH_), 256, 0, stream>>>(attn_sm, qkv_dw, qkv_raw);

    // 7) projection: out = proj_w @ attn_out   (WMMA bf16)
    gemm_bf16_wmma<<<dim3(HW_ / GN_, C_ / 16, B_), 128, 0, stream>>>(
        proj_w, qkv_raw, out, C_, C_, HW_);
}
